// pLDDTHead_36575941493396
// MI455X (gfx1250) — matrix operand order repaired and checked by reading, not compiled, not run on hardware
//
#include <hip/hip_runtime.h>

typedef __attribute__((ext_vector_type(16))) _Float16 v16h;
typedef __attribute__((ext_vector_type(8)))  _Float16 v8h;
typedef __attribute__((ext_vector_type(8)))  float    v8f;

#define S_LEN 2048
#define B_SZ  4
#define D_IN  256
#define D_H   128
#define N_B   50
#define LN_EPS 1e-5f

union V16 { v16h v; v8h p[2]; _Float16 h[16]; };

__device__ __forceinline__ void cvt4(_Float16* h, float4 f) {
  h[0] = (_Float16)f.x; h[1] = (_Float16)f.y;
  h[2] = (_Float16)f.z; h[3] = (_Float16)f.w;
}

// ---------------------------------------------------------------------------
// Pack an f32 row-major weight matrix [K x N] into f16 WMMA B-fragments.
// Fragment (kt, nt) holds the 32x16 (KxN) tile in CDNA5 B layout:
//   lane L: n = nt*16 + (L&15);  half i: k = kt*32 + (i&7) + (i>=8 ? 16 : 0) + (L>=16 ? 8 : 0)
// Stored fragment-major: dst[((kt*Ntiles + nt)*32 + lane)*16 + i]
// ---------------------------------------------------------------------------
__global__ void pack_b_kernel(const float* __restrict__ W, _Float16* __restrict__ dst,
                              int N, int Ntiles) {
  int idx  = blockIdx.x * blockDim.x + threadIdx.x;
  int i    = idx & 15;
  int lane = (idx >> 4) & 31;
  int ft   = idx >> 9;
  int nt   = ft % Ntiles;
  int kt   = ft / Ntiles;
  int k    = kt * 32 + (i & 7) + ((i >> 3) << 4) + ((lane >= 16) ? 8 : 0);
  int n    = nt * 16 + (lane & 15);
  float v  = (n < N) ? W[(size_t)k * N + n] : 0.0f;
  dst[idx] = (_Float16)v;
}

// ---------------------------------------------------------------------------
// LayerNorm(+bias) + ReLU over the D-matrix fragments of one wave, storing the
// result as f16 into an LDS row-major [16 x 128] tile (A-fragment friendly).
// acc[nt][j] holds value at row m = j + hi*8, col n = nt*16 + l16.
// ---------------------------------------------------------------------------
__device__ __forceinline__ void bias_ln_relu_store(
    v8f acc[8], const float* __restrict__ bias, const float* __restrict__ g,
    const float* __restrict__ be, _Float16* hdst, int l16, int hi) {
  float s[8], sq[8];
#pragma unroll
  for (int j = 0; j < 8; ++j) { s[j] = 0.0f; sq[j] = 0.0f; }
#pragma unroll
  for (int nt = 0; nt < 8; ++nt) {
    float bn = bias[nt * 16 + l16];
#pragma unroll
    for (int j = 0; j < 8; ++j) {
      float v = acc[nt][j] + bn;
      acc[nt][j] = v;
      s[j] += v;
      sq[j] += v * v;
    }
  }
  // reduce across the 16 lanes holding this row (width=16 keeps halves separate)
#pragma unroll
  for (int j = 0; j < 8; ++j) {
#pragma unroll
    for (int m = 1; m < 16; m <<= 1) {
      s[j]  += __shfl_xor(s[j],  m, 16);
      sq[j] += __shfl_xor(sq[j], m, 16);
    }
  }
  float mean[8], rstd[8];
#pragma unroll
  for (int j = 0; j < 8; ++j) {
    mean[j] = s[j] * (1.0f / 128.0f);
    float var = sq[j] * (1.0f / 128.0f) - mean[j] * mean[j];
    rstd[j] = rsqrtf(var + LN_EPS);
  }
#pragma unroll
  for (int nt = 0; nt < 8; ++nt) {
    int n = nt * 16 + l16;
    float gg = g[n], bb = be[n];
#pragma unroll
    for (int j = 0; j < 8; ++j) {
      float v = (acc[nt][j] - mean[j]) * rstd[j] * gg + bb;
      v = fmaxf(v, 0.0f);
      hdst[(j + hi * 8) * D_H + n] = (_Float16)v;
    }
  }
}

// ---------------------------------------------------------------------------
// Fused confidence MLP: X[8192,256] -> LN/ReLU -> [.,128] -> LN/ReLU ->
// logits[.,50]/T -> softmax.  128 threads (4 wave32) / 64 rows per block.
// ---------------------------------------------------------------------------
__global__ __launch_bounds__(128) void mlp_kernel(
    const float* __restrict__ X, const float* __restrict__ maskv,
    const _Float16* __restrict__ Wp,
    const float* __restrict__ b1, const float* __restrict__ g1, const float* __restrict__ be1,
    const float* __restrict__ b2, const float* __restrict__ g2, const float* __restrict__ be2,
    const float* __restrict__ b3, const float* __restrict__ temp,
    float* __restrict__ outLogits, float* __restrict__ outProbs) {
  __shared__ _Float16 hA[64 * D_H];   // 16 KB
  __shared__ _Float16 hB[64 * D_H];   // 16 KB

  const int tid  = threadIdx.x;
  const int wave = tid >> 5;
  const int lane = tid & 31;
  const int l16  = lane & 15;
  const int hi   = lane >> 4;
  const int rowBase = blockIdx.x * 64 + wave * 16;

  // ---------------- layer 1: X[16x256] @ W1[256x128] ----------------
  v16h A[8];
  {
    const float* xrow = X + (size_t)(rowBase + l16) * D_IN;
#pragma unroll
    for (int kc = 0; kc < 8; ++kc) {
      const float4* p0 = (const float4*)(xrow + kc * 32 + hi * 8);
      const float4* p1 = (const float4*)(xrow + kc * 32 + 16 + hi * 8);
      float4 f0 = p0[0], f1 = p0[1], f2 = p1[0], f3 = p1[1];
      V16 a;
      cvt4(a.h + 0,  f0); cvt4(a.h + 4,  f1);
      cvt4(a.h + 8,  f2); cvt4(a.h + 12, f3);
      A[kc] = a.v;
    }
  }
  v8f acc[8];
#pragma unroll
  for (int nt = 0; nt < 8; ++nt) {
    v8f c = {};
#pragma unroll
    for (int kc = 0; kc < 8; ++kc) {
      v16h b = *(const v16h*)(Wp + (size_t)(kc * 8 + nt) * 512 + lane * 16);
      c = __builtin_amdgcn_wmma_f32_16x16x32_f16(false, A[kc], false, b,
                                                 (short)0, c, false, false);
    }
    acc[nt] = c;
  }
  bias_ln_relu_store(acc, b1, g1, be1, hA + wave * 16 * D_H, l16, hi);
  __syncthreads();

  // ---------------- layer 2: h1[16x128] @ W2[128x128] ----------------
  v16h A2[4];
  {
    const _Float16* hrow = hA + (size_t)(wave * 16 + l16) * D_H;
#pragma unroll
    for (int kc = 0; kc < 4; ++kc) {
      V16 a;
      a.p[0] = *(const v8h*)(hrow + kc * 32 + hi * 8);
      a.p[1] = *(const v8h*)(hrow + kc * 32 + 16 + hi * 8);
      A2[kc] = a.v;
    }
  }
#pragma unroll
  for (int nt = 0; nt < 8; ++nt) {
    v8f c = {};
#pragma unroll
    for (int kc = 0; kc < 4; ++kc) {
      v16h b = *(const v16h*)(Wp + 32768 + (size_t)(kc * 8 + nt) * 512 + lane * 16);
      c = __builtin_amdgcn_wmma_f32_16x16x32_f16(false, A2[kc], false, b,
                                                 (short)0, c, false, false);
    }
    acc[nt] = c;
  }
  bias_ln_relu_store(acc, b2, g2, be2, hB + wave * 16 * D_H, l16, hi);
  __syncthreads();

  // ---------------- layer 3: h2[16x128] @ W3[128x64(pad)] ----------------
  v16h A3[4];
  {
    const _Float16* hrow = hB + (size_t)(wave * 16 + l16) * D_H;
#pragma unroll
    for (int kc = 0; kc < 4; ++kc) {
      V16 a;
      a.p[0] = *(const v8h*)(hrow + kc * 32 + hi * 8);
      a.p[1] = *(const v8h*)(hrow + kc * 32 + 16 + hi * 8);
      A3[kc] = a.v;
    }
  }
  float lg[4][8];
  const float invT = 1.0f / temp[0];
#pragma unroll
  for (int nt = 0; nt < 4; ++nt) {
    v8f c = {};
#pragma unroll
    for (int kc = 0; kc < 4; ++kc) {
      v16h b = *(const v16h*)(Wp + 49152 + (size_t)(kc * 4 + nt) * 512 + lane * 16);
      c = __builtin_amdgcn_wmma_f32_16x16x32_f16(false, A3[kc], false, b,
                                                 (short)0, c, false, false);
    }
    int n = nt * 16 + l16;
    float bn = (n < N_B) ? b3[n] : 0.0f;
#pragma unroll
    for (int j = 0; j < 8; ++j) lg[nt][j] = (c[j] + bn) * invT;
  }

  // ---------------- softmax over the 50 valid columns ----------------
  float mx[8];
#pragma unroll
  for (int j = 0; j < 8; ++j) mx[j] = -3.0e38f;
#pragma unroll
  for (int nt = 0; nt < 4; ++nt) {
    bool valid = (nt * 16 + l16) < N_B;
#pragma unroll
    for (int j = 0; j < 8; ++j)
      if (valid) mx[j] = fmaxf(mx[j], lg[nt][j]);
  }
#pragma unroll
  for (int j = 0; j < 8; ++j)
#pragma unroll
    for (int m = 1; m < 16; m <<= 1)
      mx[j] = fmaxf(mx[j], __shfl_xor(mx[j], m, 16));

  float ex[4][8], ssum[8];
#pragma unroll
  for (int j = 0; j < 8; ++j) ssum[j] = 0.0f;
#pragma unroll
  for (int nt = 0; nt < 4; ++nt) {
    bool valid = (nt * 16 + l16) < N_B;
#pragma unroll
    for (int j = 0; j < 8; ++j) {
      float e = valid ? __expf(lg[nt][j] - mx[j]) : 0.0f;
      ex[nt][j] = e;
      ssum[j] += e;
    }
  }
#pragma unroll
  for (int j = 0; j < 8; ++j)
#pragma unroll
    for (int m = 1; m < 16; m <<= 1)
      ssum[j] += __shfl_xor(ssum[j], m, 16);

  // ---------------- masked stores ----------------
#pragma unroll
  for (int j = 0; j < 8; ++j) {
    int r = rowBase + j + hi * 8;
    float mk = maskv[r];
    float rs = 1.0f / ssum[j];
#pragma unroll
    for (int nt = 0; nt < 4; ++nt) {
      int n = nt * 16 + l16;
      if (n < N_B) {
        outLogits[(size_t)r * N_B + n] = lg[nt][j] * mk;
        outProbs[(size_t)r * N_B + n]  = ex[nt][j] * rs * mk;
      }
    }
  }
}

// ---------------------------------------------------------------------------
// pLDDT: per batch, stage coords+mask (float4, 32 KB) in LDS, each thread owns
// one residue i and sweeps all 2048 j. Squared-distance compares (no sqrt).
// ---------------------------------------------------------------------------
__global__ __launch_bounds__(256) void plddt_kernel(
    const float* __restrict__ coords, const float* __restrict__ maskv,
    float* __restrict__ outP) {
  __shared__ float4 cj[S_LEN];  // 32 KB
  const int tid = threadIdx.x;
  const int b   = blockIdx.y;
  for (int j = tid; j < S_LEN; j += 256) {
    const float* c = coords + ((size_t)b * S_LEN + j) * 3;
    cj[j] = make_float4(c[0], c[1], c[2], maskv[(size_t)b * S_LEN + j]);
  }
  __syncthreads();

  const int i = blockIdx.x * 256 + tid;
  float4 ci = cj[i];
  float cnt = 0.0f, w0 = 0.0f, w1 = 0.0f, w2 = 0.0f, w3 = 0.0f;
#pragma unroll 4
  for (int j = 0; j < S_LEN; ++j) {
    float4 c = cj[j];
    float dx = ci.x - c.x, dy = ci.y - c.y, dz = ci.z - c.z;
    float d2 = dx * dx + dy * dy + dz * dz;
    float lj = ((d2 <= 225.0f) && (c.w > 0.0f)) ? 1.0f : 0.0f;
    cnt += lj;
    w0 += (d2 <= 0.25f) ? lj : 0.0f;
    w1 += (d2 <= 1.0f)  ? lj : 0.0f;
    w2 += (d2 <= 4.0f)  ? lj : 0.0f;
    w3 += (d2 <= 16.0f) ? lj : 0.0f;
  }
  float frac = (w0 + w1 + w2 + w3) / fmaxf(cnt, 1.0f);
  float pl = (cnt > 0.0f) ? frac * 25.0f : 0.0f;   // /4 * 100
  outP[(size_t)b * S_LEN + i] = pl * ci.w;
}

// ---------------------------------------------------------------------------
extern "C" void kernel_launch(void* const* d_in, const int* in_sizes, int n_in,
                              void* d_out, int out_size, void* d_ws, size_t ws_size,
                              hipStream_t stream) {
  (void)in_sizes; (void)n_in; (void)out_size; (void)ws_size;
  const float* X      = (const float*)d_in[0];
  const float* coords = (const float*)d_in[1];
  const float* maskv  = (const float*)d_in[2];
  const float* W1 = (const float*)d_in[3];
  const float* b1 = (const float*)d_in[4];
  const float* g1 = (const float*)d_in[5];
  const float* be1 = (const float*)d_in[6];
  const float* W2 = (const float*)d_in[7];
  const float* b2 = (const float*)d_in[8];
  const float* g2 = (const float*)d_in[9];
  const float* be2 = (const float*)d_in[10];
  const float* W3 = (const float*)d_in[11];
  const float* b3 = (const float*)d_in[12];
  const float* temp = (const float*)d_in[13];

  _Float16* Wp = (_Float16*)d_ws;
  // W1: Kt=8, Nt=8 -> 32768 halves @ 0 ; W2: Kt=4, Nt=8 -> 16384 @ 32768 ;
  // W3: Kt=4, Nt=4 (N=50 padded to 64) -> 8192 @ 49152
  pack_b_kernel<<<(8 * 8 * 512) / 256, 256, 0, stream>>>(W1, Wp,         128, 8);
  pack_b_kernel<<<(4 * 8 * 512) / 256, 256, 0, stream>>>(W2, Wp + 32768, 128, 8);
  pack_b_kernel<<<(4 * 4 * 512) / 256, 256, 0, stream>>>(W3, Wp + 49152,  50, 4);

  float* out      = (float*)d_out;
  float* outP     = out;                       // [B*S]
  float* outLogit = out + (size_t)B_SZ * S_LEN;             // [B*S*50]
  float* outProb  = out + (size_t)B_SZ * S_LEN * (1 + N_B); // [B*S*50]

  mlp_kernel<<<128, 128, 0, stream>>>(X, maskv, Wp,
                                      b1, g1, be1, b2, g2, be2, b3, temp,
                                      outLogit, outProb);
  plddt_kernel<<<dim3(S_LEN / 256, B_SZ), 256, 0, stream>>>(coords, maskv, outP);
}
